// MSA2Pair_10668698763286
// MI455X (gfx1250) — matrix operand order repaired and checked by reading, not compiled, not run on hardware
//
#include <hip/hip_runtime.h>
#include <hip/hip_bf16.h>

#define D_MSAC 256
#define D_PAIRC 128
#define D_HIDC 32
#define N_SEQ 128
#define L_RES 256
#define EPSC 1e-5f

typedef __attribute__((ext_vector_type(16))) __bf16 bf16x16;
typedef __attribute__((ext_vector_type(8)))  __bf16 bf16x8;
typedef __attribute__((ext_vector_type(8)))  float  f32x8;

#if __has_builtin(__builtin_amdgcn_global_load_async_to_lds_b128) && \
    __has_builtin(__builtin_amdgcn_s_wait_asynccnt)
#define USE_ASYNC_LDS 1
// Parameter type per hipcc diagnostic: pointer to `int vector_size(16)` in
// AS(1) (printed as `__device__`) / AS(3) for the LDS side.
typedef int v4i_ __attribute__((vector_size(16)));
typedef __attribute__((address_space(1))) v4i_ g_v4i;
typedef __attribute__((address_space(3))) v4i_ l_v4i;
#else
#define USE_ASYNC_LDS 0
#endif

__device__ inline f32x8 zero8() {
  f32x8 z;
#pragma unroll
  for (int i = 0; i < 8; ++i) z[i] = 0.0f;
  return z;
}

__device__ inline bf16x16 join8(bf16x8 lo, bf16x8 hi) {
  bf16x16 r;
#pragma unroll
  for (int e = 0; e < 8; ++e) { r[e] = lo[e]; r[e + 8] = hi[e]; }
  return r;
}

// ---------------------------------------------------------------------------
// Kernel 0: Wo [1024 x 128] f32  ->  WoT [128 x 1024] bf16 (column-major pack)
// ---------------------------------------------------------------------------
__global__ void wo_pack_kernel(const float* __restrict__ Wo,
                               __bf16* __restrict__ WoT) {
  int idx = blockIdx.x * blockDim.x + threadIdx.x;   // 0 .. 131071
  int k = idx >> 7;       // 0..1023
  int p = idx & 127;      // 0..127
  WoT[(size_t)p * 1024 + k] = (__bf16)Wo[idx];
}

// ---------------------------------------------------------------------------
// Kernel 1: LayerNorm + dual projection.
//  One wave (32 lanes) per (s,l) row. Writes transposed bf16:
//    leftT [l][i][s]          (i.e. (l*32+i)*128 + s)
//    rightT[(m*32+j)][s]      (same indexing, m==l here), scaled by 1/N
//  Weight panels staged into LDS via gfx1250 async global->LDS copies
//  (ASYNCcnt path) when available, else plain ds stores.
// ---------------------------------------------------------------------------
__device__ inline void stage_weights(const float* __restrict__ W,
                                     float* sW, int tid) {
#if USE_ASYNC_LDS
#pragma unroll
  for (int r = 0; r < 8; ++r) {
    float* src = const_cast<float*>(W) + r * 1024 + tid * 4;   // 16B per lane
    __builtin_amdgcn_global_load_async_to_lds_b128(
        (g_v4i*)src, (l_v4i*)(sW + r * 1024 + tid * 4), 0, 0);
  }
  __builtin_amdgcn_s_wait_asynccnt(0);
#else
#pragma unroll
  for (int r = 0; r < 32; ++r) sW[tid + r * 256] = W[tid + r * 256];
#endif
}

__global__ void ln_proj_kernel(const float* __restrict__ msa,
                               const float* __restrict__ gamma,
                               const float* __restrict__ beta,
                               const float* __restrict__ Wl,
                               const float* __restrict__ bl,
                               const float* __restrict__ Wr,
                               const float* __restrict__ br,
                               __bf16* __restrict__ leftT,
                               __bf16* __restrict__ rightT) {
  __shared__ float sW[D_MSAC * D_HIDC];   // 32 KB, holds Wl then Wr
  __shared__ float sX[8 * D_MSAC];        // 8 KB, normalized rows (1 per wave)

  const int tid  = threadIdx.x;
  const int wave = tid >> 5;
  const int lane = tid & 31;

  stage_weights(Wl, sW, tid);

  const int row = blockIdx.x * 8 + wave;   // over N*L = 32768
  const int s   = row >> 8;                // / L_RES
  const int l   = row & (L_RES - 1);
  const float* xrow = msa + (size_t)row * D_MSAC;

  float xv[8];
  float sum = 0.0f, sq = 0.0f;
#pragma unroll
  for (int q = 0; q < 8; ++q) {
    xv[q] = xrow[q * 32 + lane];
    sum += xv[q];
    sq  += xv[q] * xv[q];
  }
  // wave32 butterfly reduction
#pragma unroll
  for (int off = 16; off > 0; off >>= 1) {
    sum += __shfl_xor(sum, off, 32);
    sq  += __shfl_xor(sq,  off, 32);
  }
  const float mu   = sum * (1.0f / D_MSAC);
  const float var  = sq * (1.0f / D_MSAC) - mu * mu;
  const float rstd = rsqrtf(var + EPSC);
#pragma unroll
  for (int q = 0; q < 8; ++q) {
    const int d = q * 32 + lane;
    sX[wave * D_MSAC + d] = (xv[q] - mu) * rstd * gamma[d] + beta[d];
  }
  __syncthreads();

  // left = xn @ Wl + bl    (lane = output channel)
  float acc = bl[lane];
  for (int d = 0; d < D_MSAC; ++d)
    acc = fmaf(sX[wave * D_MSAC + d], sW[d * 32 + lane], acc);
  leftT[((size_t)l * 32 + lane) * 128 + s] = (__bf16)acc;

  __syncthreads();
  stage_weights(Wr, sW, tid);
  __syncthreads();

  float accr = br[lane];
  for (int d = 0; d < D_MSAC; ++d)
    accr = fmaf(sX[wave * D_MSAC + d], sW[d * 32 + lane], accr);
  rightT[((size_t)l * 32 + lane) * 128 + s] = (__bf16)(accr * (1.0f / N_SEQ));
}

// ---------------------------------------------------------------------------
// Kernel 2: fused outer-product (GEMM1) + output projection (GEMM2) + residual
//  Workgroup = 8 waves, handles (l = blockIdx.y, m-tile16 = blockIdx.x).
//  GEMM1: leftT_l^T [32i x 128s] @ right [128s x (16m*32j)] -> C1, bf16 -> LDS
//  GEMM2: A2[16m x 1024(i*32+j)] @ WoT -> [16m x 128p], + pair + bo
// ---------------------------------------------------------------------------
__global__ void opm_proj_kernel(const __bf16* __restrict__ leftT,
                                const __bf16* __restrict__ rightT,
                                const __bf16* __restrict__ WoT,
                                const float* __restrict__ pair,
                                const float* __restrict__ bo,
                                float* __restrict__ out) {
  __shared__ __bf16 sA2[16 * 1024];   // 32 KB: C1 block as bf16, [m_local][i*32+j]

  const int tid  = threadIdx.x;
  const int wave = tid >> 5;
  const int lane = tid & 31;
  const int half = lane >> 4;     // 0: lanes 0-15, 1: lanes 16-31
  const int lr   = lane & 15;

  const int mt = blockIdx.x;      // m-tile (16 m's)
  const int l  = blockIdx.y;      // residue index

  // warm the L2/L0 path for the epilogue residual tile (global_prefetch_b8)
  const float* pairTile =
      pair + ((size_t)l * L_RES + (size_t)mt * 16) * D_PAIRC;
  __builtin_prefetch(pairTile + tid * 8, 0, 1);

  f32x8 acc[8];
#pragma unroll
  for (int t = 0; t < 8; ++t) acc[t] = zero8();

  const __bf16* Abase = leftT + (size_t)l * (32 * 128);

  // ---- GEMM1: K = s (128), step 32 -------------------------------------
  for (int ks = 0; ks < 128; ks += 32) {
    // A fragments (16x32 bf16): lane holds row (lane&15), K-halves split by lane>=16
    bf16x16 afr[2];
#pragma unroll
    for (int it = 0; it < 2; ++it) {
      const __bf16* ap = Abase + (size_t)(it * 16 + lr) * 128 + ks + half * 8;
      afr[it] = join8(*(const bf16x8*)ap, *(const bf16x8*)(ap + 16));
    }
    // Preload ALL four B fragments first so the loads land in one clause
    // and the 8 WMMAs below issue back-to-back.
    bf16x16 bfr[4];
#pragma unroll
    for (int q = 0; q < 4; ++q) {
      const int nt = wave * 4 + q;          // n-tile 0..31 over (m_local, j)
      const int n  = nt * 16 + lr;          // this lane's B column
      const int m  = mt * 16 + (n >> 5);    // global m
      const int j  = n & 31;
      const __bf16* bp = rightT + ((size_t)m * 32 + j) * 128 + ks + half * 8;
      bfr[q] = join8(*(const bf16x8*)bp, *(const bf16x8*)(bp + 16));
    }
#pragma unroll
    for (int q = 0; q < 4; ++q)
#pragma unroll
      for (int it = 0; it < 2; ++it)
        acc[it * 4 + q] = __builtin_amdgcn_wmma_f32_16x16x32_bf16(
            false, afr[it], false, bfr[q], (short)0, acc[it * 4 + q],
            false, false);
  }

  // ---- C1 (f32) -> bf16 into LDS as A2[m_local][i*32+j] ----------------
#pragma unroll
  for (int q = 0; q < 4; ++q) {
    const int nt   = wave * 4 + q;
    const int mloc = nt >> 1;
    const int j    = (nt & 1) * 16 + lr;    // C/D: N = lane&15
#pragma unroll
    for (int it = 0; it < 2; ++it) {
#pragma unroll
      for (int v = 0; v < 8; ++v) {
        const int i = it * 16 + v + half * 8;   // C/D: M = v + 8*(lane>=16)
        sA2[mloc * 1024 + i * 32 + j] = (__bf16)acc[it * 4 + q][v];
      }
    }
  }
  __syncthreads();

  // ---- GEMM2: [16m x 1024] @ WoT -> wave's 16x16 p-tile ----------------
  // Two independent accumulator chains so WMMAs from chain A hide chain B's
  // fragment loads (and vice versa).
  f32x8 accA = zero8();
  f32x8 accB = zero8();
  const int p0 = wave * 16;
  const __bf16* WoCol = WoT + (size_t)(p0 + lr) * 1024;
#pragma unroll 4
  for (int k0 = 0; k0 < 1024; k0 += 64) {
    const __bf16* ap0 = &sA2[lr * 1024 + k0 + half * 8];
    bf16x16 a0 = join8(*(const bf16x8*)ap0, *(const bf16x8*)(ap0 + 16));
    const __bf16* bp0 = WoCol + k0 + half * 8;
    bf16x16 b0 = join8(*(const bf16x8*)bp0, *(const bf16x8*)(bp0 + 16));

    const __bf16* ap1 = &sA2[lr * 1024 + (k0 + 32) + half * 8];
    bf16x16 a1 = join8(*(const bf16x8*)ap1, *(const bf16x8*)(ap1 + 16));
    const __bf16* bp1 = WoCol + (k0 + 32) + half * 8;
    bf16x16 b1 = join8(*(const bf16x8*)bp1, *(const bf16x8*)(bp1 + 16));

    accA = __builtin_amdgcn_wmma_f32_16x16x32_bf16(
        false, a0, false, b0, (short)0, accA, false, false);
    accB = __builtin_amdgcn_wmma_f32_16x16x32_bf16(
        false, a1, false, b1, (short)0, accB, false, false);
  }
  f32x8 acc2;
#pragma unroll
  for (int v = 0; v < 8; ++v) acc2[v] = accA[v] + accB[v];

  // ---- epilogue: out = pair + C2 + bo ----------------------------------
#pragma unroll
  for (int v = 0; v < 8; ++v) {
    const int mrow = v + half * 8;
    const int m = mt * 16 + mrow;
    const int p = p0 + lr;
    const size_t o = ((size_t)l * L_RES + m) * D_PAIRC + p;
    out[o] = pair[o] + acc2[v] + bo[p];
  }
}

// ---------------------------------------------------------------------------
extern "C" void kernel_launch(void* const* d_in, const int* in_sizes, int n_in,
                              void* d_out, int out_size, void* d_ws, size_t ws_size,
                              hipStream_t stream) {
  const float* msa   = (const float*)d_in[0];
  const float* pair  = (const float*)d_in[1];
  const float* gamma = (const float*)d_in[2];
  const float* beta  = (const float*)d_in[3];
  const float* Wl    = (const float*)d_in[4];
  const float* bl    = (const float*)d_in[5];
  const float* Wr    = (const float*)d_in[6];
  const float* br    = (const float*)d_in[7];
  const float* Wo    = (const float*)d_in[8];
  const float* bo    = (const float*)d_in[9];
  float* out = (float*)d_out;

  char* ws = (char*)d_ws;
  __bf16* leftT  = (__bf16*)(ws);                         // 2 MB
  __bf16* rightT = (__bf16*)(ws + (2u << 20));            // 2 MB
  __bf16* WoT    = (__bf16*)(ws + (4u << 20));            // 256 KB

  wo_pack_kernel<<<512, 256, 0, stream>>>(Wo, WoT);
  ln_proj_kernel<<<4096, 256, 0, stream>>>(msa, gamma, beta, Wl, bl, Wr, br,
                                           leftT, rightT);
  dim3 g2(16, L_RES);
  opm_proj_kernel<<<g2, 256, 0, stream>>>(leftT, rightT, WoT, pair, bo, out);
}
